// CrossMatrix_68573447848140
// MI455X (gfx1250) — compile-verified, matrix operand order checked by reading, not run on hardware
//
#include <hip/hip_runtime.h>
#include <hip/hip_bf16.h>

typedef __attribute__((ext_vector_type(16))) _Float16 v16h;
typedef __attribute__((ext_vector_type(8)))  _Float16 v8h;
typedef __attribute__((ext_vector_type(8)))  float    v8f;
typedef int vi4 __attribute__((vector_size(16)));

#define B_ROWS 65536
#define NP1    65
#define INNER  4225          // 65*65
#define QPAD   80            // q padded 65 -> 80 (multiple of 16)
#define KBLK   32
#define CHUNK_KB 4           // K-blocks per LDS staging chunk
#define NCHUNK 42            // chunks
#define NKB    (NCHUNK * CHUNK_KB)   // 168 K-blocks (K_pad = 5376, zero-padded)
#define CHUNK_BYTES (CHUNK_KB * 4 * 32 * 32)  // 4 kb * 4 tiles * 32 lanes * 32B = 16384
#define EPS    1e-6f

#if __has_builtin(__builtin_amdgcn_global_load_async_to_lds_b128) && \
    __has_builtin(__builtin_amdgcn_s_wait_asynccnt)
#define USE_ASYNC 1
#else
#define USE_ASYNC 0
#endif

#define ASG __attribute__((address_space(1)))
#define ASL __attribute__((address_space(3)))

// ---------------------------------------------------------------------------
// Prep: pack W [64,4225] f32 into f16 WMMA B-fragment stream (zero padded).
//   frag(kb,t): 32 lanes x 16 f16 (32B/lane contiguous).
//   lane l, elem i -> K = kb*32 + (l>=16?16:0) + i ; n = t*16 + (l&15)
//   K -> p = K/80, q = K%80 ; value = W[n*4225 + p*65 + q] if p,q < 65 else 0
// ---------------------------------------------------------------------------
__global__ void pack_w_kernel(const float* __restrict__ W, _Float16* __restrict__ Wp) {
    int tid = blockIdx.x * blockDim.x + threadIdx.x;    // one per f16 pair
    if (tid >= NKB * 4 * 32 * 8) return;
    int v    = tid & 7;
    int lane = (tid >> 3) & 31;
    int t    = (tid >> 8) & 3;
    int kb   = tid >> 10;
    int n     = t * 16 + (lane & 15);
    int kbase = kb * 32 + ((lane >> 4) << 4) + 2 * v;
#pragma unroll
    for (int e = 0; e < 2; ++e) {
        int K = kbase + e;
        int p = K / QPAD;
        int q = K - p * QPAD;
        float val = 0.0f;
        if (p < NP1 && q < NP1) val = W[n * INNER + p * NP1 + q];
        Wp[tid * 2 + e] = (_Float16)val;
    }
}

// ---------------------------------------------------------------------------
// Main: 256 threads = 8 waves; each wave owns a 16(batch) x 64(N) tile.
//   A[row,K=(p,q)] = x1[row,p]*x1[row,q] built from f16 x1 rows in LDS.
//   B fragments double-buffer staged into LDS via async global->LDS DMA.
//   672 v_wmma_f32_16x16x32_f16 per wave, f32 accumulation, LN in registers.
// ---------------------------------------------------------------------------
__global__ __launch_bounds__(256) void crossmatrix_kernel(
    const float* __restrict__ x, const _Float16* __restrict__ Wp,
    const float* __restrict__ bias, const float* __restrict__ gamma,
    const float* __restrict__ beta, float* __restrict__ out)
{
    __shared__ __align__(32) char     ldsB[2][CHUNK_BYTES];   // 32 KB
    __shared__ __align__(16) _Float16 x1h[8][16][QPAD];       // 20 KB
    const int tid  = threadIdx.x;
    const int lane = tid & 31;
    const int wave = tid >> 5;
    const int r    = lane & 15;    // A-matrix row (M = lane%16)
    const int h    = lane >> 4;    // lane half: K-offsets 0..15 / 16..31
    const int row0 = blockIdx.x * 128 + wave * 16;

    // ---- stage x1 rows into LDS as f16: [1, x[0..63], zeros..79] ----
    {
        const float* src = x + (size_t)(row0 + r) * 64 + h * 32;
        _Float16* dstrow = &x1h[wave][r][0];
        if (h == 0) dstrow[0] = (_Float16)1.0f;
#pragma unroll
        for (int i = 0; i < 8; ++i) {
            float4 v = ((const float4*)src)[i];
            _Float16* d = dstrow + 1 + h * 32 + i * 4;
            d[0] = (_Float16)v.x; d[1] = (_Float16)v.y;
            d[2] = (_Float16)v.z; d[3] = (_Float16)v.w;
        }
        if (h == 1) {
#pragma unroll
            for (int i = NP1; i < QPAD; ++i) dstrow[i] = (_Float16)0.0f;
        }
    }

    // ---- B chunk staging: 16 KB per chunk, 256 threads x 4 x 16B ----
    auto stage = [&](int c, int buf) {
        const char* g = (const char*)Wp + (size_t)c * CHUNK_BYTES;
        char* l = &ldsB[buf][0];
#pragma unroll
        for (int j = 0; j < 4; ++j) {
            int off = (j * 256 + tid) * 16;
#if USE_ASYNC
            __builtin_amdgcn_global_load_async_to_lds_b128(
                (ASG vi4*)(g + off), (ASL vi4*)(l + off), 0, 0);
#else
            *(float4*)(l + off) = *(const float4*)(g + off);
#endif
        }
    };

    stage(0, 0);

    v8f acc0 = {}, acc1 = {}, acc2 = {}, acc3 = {};
    const _Float16* xrowh = &x1h[wave][r][0];
    int p = 0;          // outer index for this lane's 16-K window
    int qw = h * 16;    // window start in padded q row (multiple of 16)

    for (int c = 0; c < NCHUNK; ++c) {
        if (c + 1 < NCHUNK) {
            stage(c + 1, (c + 1) & 1);
#if USE_ASYNC
            __builtin_amdgcn_s_wait_asynccnt(4);   // chunk c complete, c+1 in flight
#endif
        } else {
#if USE_ASYNC
            __builtin_amdgcn_s_wait_asynccnt(0);
#endif
        }
        __syncthreads();
        const char* bbase = &ldsB[c & 1][0];
#pragma unroll
        for (int kk = 0; kk < CHUNK_KB; ++kk) {
            // ---- A fragment: a[i] = x1[p] * x1[qw+i]  (f16 packed muls) ----
            _Float16 xph = xrowh[p];
            v8h xq0 = *(const v8h*)(xrowh + qw);
            v8h xq1 = *(const v8h*)(xrowh + qw + 8);
            v8h pr0 = xph * xq0;
            v8h pr1 = xph * xq1;
            v16h a = __builtin_shufflevector(pr0, pr1,
                        0,1,2,3,4,5,6,7,8,9,10,11,12,13,14,15);
            // ---- B fragments from LDS ----
            const v16h* bf = (const v16h*)(bbase + kk * 4096) + lane;
            v16h b0 = bf[0];
            v16h b1 = bf[32];
            v16h b2 = bf[64];
            v16h b3 = bf[96];
            acc0 = __builtin_amdgcn_wmma_f32_16x16x32_f16(false, a, false, b0, (short)0, acc0, false, false);
            acc1 = __builtin_amdgcn_wmma_f32_16x16x32_f16(false, a, false, b1, (short)0, acc1, false, false);
            acc2 = __builtin_amdgcn_wmma_f32_16x16x32_f16(false, a, false, b2, (short)0, acc2, false, false);
            acc3 = __builtin_amdgcn_wmma_f32_16x16x32_f16(false, a, false, b3, (short)0, acc3, false, false);
            // ---- advance (p, qw): +32 within an 80-wide padded q row ----
            qw += 32;
            if (qw >= QPAD) { qw -= QPAD; ++p; }
        }
        __syncthreads();   // all waves done with buf (c&1) before re-staging it
    }

    // ---- epilogue: bias + LayerNorm over N=64, all in registers ----
    // C layout: VGPR reg, lane-half hh -> row = reg + 8*hh ; col = t*16 + lane%16
    const int nloc = lane & 15;
    float bb[4], gg[4], be[4];
#pragma unroll
    for (int t = 0; t < 4; ++t) {
        bb[t] = bias [t * 16 + nloc];
        gg[t] = gamma[t * 16 + nloc];
        be[t] = beta [t * 16 + nloc];
    }
    float* outp = out + (size_t)row0 * 64;
#pragma unroll
    for (int reg = 0; reg < 8; ++reg) {
        float y0 = acc0[reg] + bb[0];
        float y1 = acc1[reg] + bb[1];
        float y2 = acc2[reg] + bb[2];
        float y3 = acc3[reg] + bb[3];
        float s  = y0 + y1 + y2 + y3;
        float ss = y0 * y0 + y1 * y1 + y2 * y2 + y3 * y3;
#pragma unroll
        for (int m = 8; m; m >>= 1) {        // reduce within 16-lane half
            s  += __shfl_xor(s,  m, 32);
            ss += __shfl_xor(ss, m, 32);
        }
        float u   = s * (1.0f / 64.0f);
        float var = ss * (1.0f / 64.0f) - u * u;
        float inv = rsqrtf(var + EPS);
        int rowo = reg + h * 8;
        float* po = outp + (size_t)rowo * 64 + nloc;
        po[0]  = gg[0] * (y0 - u) * inv + be[0];
        po[16] = gg[1] * (y1 - u) * inv + be[1];
        po[32] = gg[2] * (y2 - u) * inv + be[2];
        po[48] = gg[3] * (y3 - u) * inv + be[3];
    }
}

extern "C" void kernel_launch(void* const* d_in, const int* in_sizes, int n_in,
                              void* d_out, int out_size, void* d_ws, size_t ws_size,
                              hipStream_t stream) {
    const float* x     = (const float*)d_in[0];
    const float* W     = (const float*)d_in[1];
    const float* bias  = (const float*)d_in[2];
    const float* gamma = (const float*)d_in[3];
    const float* beta  = (const float*)d_in[4];
    float* out = (float*)d_out;
    _Float16* Wp = (_Float16*)d_ws;   // needs NKB*4096 = 688,128 B of scratch

    int prepN = NKB * 4 * 32 * 8;     // one thread per f16 pair
    pack_w_kernel<<<(prepN + 255) / 256, 256, 0, stream>>>(W, Wp);
    crossmatrix_kernel<<<B_ROWS / 128, 256, 0, stream>>>(x, Wp, bias, gamma, beta, out);
}
